// MultiHeadedAttention_44994077393576
// MI455X (gfx1250) — matrix operand order repaired and checked by reading, not compiled, not run on hardware
//
#include <hip/hip_runtime.h>
#include <hip/hip_bf16.h>
#include <math.h>

typedef __attribute__((ext_vector_type(16))) __bf16 v16bf;
typedef __attribute__((ext_vector_type(8)))  float  v8f;
typedef int v4i __attribute__((vector_size(16)));

constexpr int cB  = 2;
constexpr int cS  = 2048;
constexpr int cD  = 1024;
constexpr int cH  = 16;
constexpr int cDK = 64;
constexpr int cM  = cB * cS;          // 4096 rows

// ---------------------------------------------------------------------------
// CDNA5 async global->LDS staging (ASYNCcnt-tracked), with safe fallback.
// Probe: builtin signature is (AS1 v4i*, AS3 v4i*, imm offset, imm cpol).
// ---------------------------------------------------------------------------
#define AS1 __attribute__((address_space(1)))
#define AS3 __attribute__((address_space(3)))

#if defined(__AMDGCN__) && __has_builtin(__builtin_amdgcn_global_load_async_to_lds_b128)
#define HAVE_ASYNC_LDS 1
#else
#define HAVE_ASYNC_LDS 0
#endif

static __device__ __forceinline__ void async_copy16(const void* g, void* l)
{
#if HAVE_ASYNC_LDS
    __builtin_amdgcn_global_load_async_to_lds_b128((AS1 v4i*)g, (AS3 v4i*)l, 0, 0);
#else
    *(uint4*)l = *(const uint4*)g;
#endif
}

static __device__ __forceinline__ void async_wait()
{
#if HAVE_ASYNC_LDS
#if __has_builtin(__builtin_amdgcn_s_wait_asynccnt)
    __builtin_amdgcn_s_wait_asynccnt(0);
#else
    asm volatile("s_wait_asynccnt 0x0" ::: "memory");
#endif
#endif
}

// ---------------------------------------------------------------------------
// LayerNorm (torch variant: unbiased std, eps added to std), fp32 -> bf16
// One block per row, 256 threads, 4 floats per thread.
// ---------------------------------------------------------------------------
__global__ void __launch_bounds__(256)
ln_kernel(const float* __restrict__ x, const float* __restrict__ a2,
          const float* __restrict__ b2, __bf16* __restrict__ ybf,
          float* __restrict__ yf)
{
    const int row = blockIdx.x;
    const int tid = threadIdx.x;
    const float4 xv = ((const float4*)(x + (size_t)row * cD))[tid];

    float s  = xv.x + xv.y + xv.z + xv.w;
    float ss = xv.x * xv.x + xv.y * xv.y + xv.z * xv.z + xv.w * xv.w;
    #pragma unroll
    for (int off = 16; off >= 1; off >>= 1) {
        s  += __shfl_xor(s,  off, 32);
        ss += __shfl_xor(ss, off, 32);
    }
    __shared__ float red[16];
    const int wid = tid >> 5, lane = tid & 31;
    if (lane == 0) { red[wid] = s; red[wid + 8] = ss; }
    __syncthreads();
    if (tid == 0) {
        float t = 0.f, t2 = 0.f;
        #pragma unroll
        for (int i = 0; i < 8; i++) { t += red[i]; t2 += red[i + 8]; }
        red[0] = t; red[8] = t2;
    }
    __syncthreads();
    s  = red[0];
    ss = red[8];

    const float mean = s * (1.0f / cD);
    const float var  = (ss - (float)cD * mean * mean) * (1.0f / (cD - 1));
    const float inv  = 1.0f / (sqrtf(var) + 1e-6f);

    const int c = tid * 4;
    const float4 av = ((const float4*)a2)[tid];
    const float4 bv = ((const float4*)b2)[tid];
    float y0 = av.x * (xv.x - mean) * inv + bv.x;
    float y1 = av.y * (xv.y - mean) * inv + bv.y;
    float y2 = av.z * (xv.z - mean) * inv + bv.z;
    float y3 = av.w * (xv.w - mean) * inv + bv.w;

    __bf16* yb = ybf + (size_t)row * cD + c;
    yb[0] = (__bf16)y0; yb[1] = (__bf16)y1; yb[2] = (__bf16)y2; yb[3] = (__bf16)y3;
    if (yf) {
        float4 o; o.x = y0; o.y = y1; o.z = y2; o.w = y3;
        ((float4*)(yf + (size_t)row * cD))[tid] = o;
    }
}

// ---------------------------------------------------------------------------
// fp32 -> bf16 weight conversion, 4 elems/thread
// ---------------------------------------------------------------------------
__global__ void __launch_bounds__(256)
cvt_bf16(const float* __restrict__ s, __bf16* __restrict__ d)
{
    const int i = (blockIdx.x * 256 + threadIdx.x) * 4;
    const float4 f = *(const float4*)(s + i);
    d[i + 0] = (__bf16)f.x; d[i + 1] = (__bf16)f.y;
    d[i + 2] = (__bf16)f.z; d[i + 3] = (__bf16)f.w;
}

// ---------------------------------------------------------------------------
// Fragment builders (wave32 WMMA 16x16x32 bf16 layouts):
//  A-frag: lane lo = M row; elem e -> K = (e<8? e : e+8) + half*8
//          -> two contiguous 8-elem chunks at K = half*8 and half*8+16
//  B-frag: lane lo = N col; elem e -> K = (e<8? e : e+8) + half*8
//          -> two contiguous 8-elem chunks at K = half*8 and half*8+16 (needs K-major LDS)
//  C-frag: lane lo = N col; reg r  -> M = r + half*8
// ---------------------------------------------------------------------------
static __device__ __forceinline__ v16bf load_afrag(const __bf16* base /* K-contig row */,
                                                   int half)
{
    v16bf a;
    const __bf16* p = base + half * 8;
    ((uint4*)&a)[0] = *(const uint4*)(p);
    ((uint4*)&a)[1] = *(const uint4*)(p + 16);
    return a;
}

static __device__ __forceinline__ v16bf load_bfrag(const __bf16* base /* K-contig col */,
                                                   int half)
{
    v16bf b;
    const __bf16* p = base + half * 8;
    ((uint4*)&b)[0] = *(const uint4*)(p);
    ((uint4*)&b)[1] = *(const uint4*)(p + 16);
    return b;
}

// ---------------------------------------------------------------------------
// bf16 GEMM (M=4096, N=K=1024), C = A*W + bias.
// Block tile 128x64, BK=32. 256 threads = 8 waves; wave w owns rows w*16..+15.
// A staged direct (async b128), W staged K-major (BsT[n][k]) so B-frags are
// two ds_load_b128.
// ---------------------------------------------------------------------------
__global__ void __launch_bounds__(256)
gemm16_qkv(const __bf16* __restrict__ A, const __bf16* __restrict__ W,
           const float* __restrict__ bias, __bf16* __restrict__ outBHSD)
{
    constexpr int BM = 128, BN = 64, BK = 32, K = cD, N = cD;
    __shared__ __align__(16) __bf16 As[BM * BK];
    __shared__ __align__(16) __bf16 BsT[BN * BK];   // [n][k]

    const int tid = threadIdx.x;
    const int m0 = blockIdx.x * BM, n0 = blockIdx.y * BN;
    const int wid = tid >> 5, lane = tid & 31, lo = lane & 15, half = lane >> 4;

    v8f z = {0.f,0.f,0.f,0.f,0.f,0.f,0.f,0.f};
    v8f acc[4];
    #pragma unroll
    for (int i = 0; i < 4; i++) acc[i] = z;

    for (int k0 = 0; k0 < K; k0 += BK) {
        // stage W tile transposed: read 8 bf16 along N, scatter along K-major
        {
            const int row = tid >> 3, q = tid & 7;     // row: k, q: n-chunk
            uint4 t = *(const uint4*)(W + (size_t)(k0 + row) * N + n0 + q * 8);
            const __bf16* tb = (const __bf16*)&t;
            #pragma unroll
            for (int jj = 0; jj < 8; jj++)
                BsT[(q * 8 + jj) * BK + row] = tb[jj];
            if (k0 + BK < K)
                __builtin_prefetch(W + (size_t)(k0 + BK + row) * N + n0 + q * 8, 0, 1);
        }
        // stage A tile: 512 x b128 async copies
        #pragma unroll
        for (int t = 0; t < 2; t++) {
            const int i = tid + t * 256;
            const int row = i >> 2, q = i & 3;
            async_copy16(A + (size_t)(m0 + row) * K + k0 + q * 8,
                         (uint4*)As + i);
        }
        async_wait();
        __syncthreads();

        const v16bf a = load_afrag(As + (wid * 16 + lo) * BK, half);
        #pragma unroll
        for (int nt = 0; nt < 4; nt++) {
            const v16bf bfr = load_bfrag(BsT + (nt * 16 + lo) * BK, half);
            acc[nt] = __builtin_amdgcn_wmma_f32_16x16x32_bf16(
                false, a, false, bfr, (short)0, acc[nt], false, false);
        }
        __syncthreads();
    }

    #pragma unroll
    for (int nt = 0; nt < 4; nt++)
        #pragma unroll
        for (int r = 0; r < 8; r++) {
            const int row = m0 + wid * 16 + r + half * 8;
            const int col = n0 + nt * 16 + lo;
            const float v = acc[nt][r] + bias[col];
            const int b = row / cS, s = row % cS;
            const int h = col >> 6, dk = col & 63;
            outBHSD[((size_t)((b * cH + h) * cS + s)) * cDK + dk] = (__bf16)v;
        }
}

// Same GEMM, fp32 epilogue with bias + residual (normalized v)
__global__ void __launch_bounds__(256)
gemm16_out(const __bf16* __restrict__ A, const __bf16* __restrict__ W,
           const float* __restrict__ bias, const float* __restrict__ vres,
           float* __restrict__ outp)
{
    constexpr int BM = 128, BN = 64, BK = 32, K = cD, N = cD;
    __shared__ __align__(16) __bf16 As[BM * BK];
    __shared__ __align__(16) __bf16 BsT[BN * BK];

    const int tid = threadIdx.x;
    const int m0 = blockIdx.x * BM, n0 = blockIdx.y * BN;
    const int wid = tid >> 5, lane = tid & 31, lo = lane & 15, half = lane >> 4;

    v8f z = {0.f,0.f,0.f,0.f,0.f,0.f,0.f,0.f};
    v8f acc[4];
    #pragma unroll
    for (int i = 0; i < 4; i++) acc[i] = z;

    for (int k0 = 0; k0 < K; k0 += BK) {
        {
            const int row = tid >> 3, q = tid & 7;
            uint4 t = *(const uint4*)(W + (size_t)(k0 + row) * N + n0 + q * 8);
            const __bf16* tb = (const __bf16*)&t;
            #pragma unroll
            for (int jj = 0; jj < 8; jj++)
                BsT[(q * 8 + jj) * BK + row] = tb[jj];
        }
        #pragma unroll
        for (int t = 0; t < 2; t++) {
            const int i = tid + t * 256;
            const int row = i >> 2, q = i & 3;
            async_copy16(A + (size_t)(m0 + row) * K + k0 + q * 8,
                         (uint4*)As + i);
        }
        async_wait();
        __syncthreads();

        const v16bf a = load_afrag(As + (wid * 16 + lo) * BK, half);
        #pragma unroll
        for (int nt = 0; nt < 4; nt++) {
            const v16bf bfr = load_bfrag(BsT + (nt * 16 + lo) * BK, half);
            acc[nt] = __builtin_amdgcn_wmma_f32_16x16x32_bf16(
                false, a, false, bfr, (short)0, acc[nt], false, false);
        }
        __syncthreads();
    }

    #pragma unroll
    for (int nt = 0; nt < 4; nt++)
        #pragma unroll
        for (int r = 0; r < 8; r++) {
            const int row = m0 + wid * 16 + r + half * 8;
            const int col = n0 + nt * 16 + lo;
            const size_t idx = (size_t)row * cD + col;
            outp[idx] = acc[nt][r] + bias[col] + vres[idx];
        }
}

// ---------------------------------------------------------------------------
// Flash-style attention: one block per (b, h, 64 query rows); 4 waves,
// wave w owns q rows [qTile + 16w, +16). Online softmax, never materializes
// the SxS score matrix. Q/K/V are bf16 [B,H,S,DK]; X out is bf16 [B,S,D].
// K tile stored [key][dk] (dk-contig -> score B-frags are b128 loads);
// V tile stored transposed [dk][key] (key-contig -> PV B-frags are b128).
// ---------------------------------------------------------------------------
__global__ void __launch_bounds__(128)
attn_fused(const __bf16* __restrict__ Q, const __bf16* __restrict__ Kg,
           const __bf16* __restrict__ Vg, const int* __restrict__ mask,
           __bf16* __restrict__ X)
{
    __shared__ __align__(16) __bf16 Ks[64 * 64];        // [key][dk]
    __shared__ __align__(16) __bf16 VsT[64 * 64];       // [dk][key]
    __shared__ __align__(16) __bf16 Ps[4][16 * 64];     // per-wave P pad

    const int tid = threadIdx.x, wid = tid >> 5, lane = tid & 31;
    const int lo = lane & 15, half = lane >> 4;
    const int b = blockIdx.z, h = blockIdx.y;
    const int qTile = blockIdx.x * 64;
    const size_t bh = (size_t)(b * cH + h) * cS * cDK;
    const __bf16* Qp = Q + bh;
    const __bf16* Kp = Kg + bh;
    const __bf16* Vp = Vg + bh;
    const int* mp = mask + (size_t)b * cS * cS;
    const int qw = qTile + wid * 16;

    // Q fragments for this wave's 16 rows (dk 0..31 and 32..63)
    v16bf aq[2];
    #pragma unroll
    for (int j = 0; j < 2; j++) {
        const __bf16* qp = Qp + (size_t)(qw + lo) * cDK + j * 32 + half * 8;
        ((uint4*)&aq[j])[0] = *(const uint4*)(qp);
        ((uint4*)&aq[j])[1] = *(const uint4*)(qp + 16);
    }

    v8f z = {0.f,0.f,0.f,0.f,0.f,0.f,0.f,0.f};
    v8f o[4];
    #pragma unroll
    for (int i = 0; i < 4; i++) o[i] = z;
    float mrow[8], lrow[8];
    #pragma unroll
    for (int r = 0; r < 8; r++) { mrow[r] = -1e30f; lrow[r] = 0.f; }

    for (int kt = 0; kt < cS / 64; ++kt) {
        // stage K tile direct (async b128), V tile transposed (scatter b16)
        #pragma unroll
        for (int t = 0; t < 4; t++) {
            const int i = tid + t * 128;
            const int row = i >> 3, q = i & 7;
            async_copy16(Kp + (size_t)(kt * 64 + row) * cDK + q * 8,
                         Ks + row * 64 + q * 8);
            uint4 tv = *(const uint4*)(Vp + (size_t)(kt * 64 + row) * cDK + q * 8);
            const __bf16* tb = (const __bf16*)&tv;
            #pragma unroll
            for (int jj = 0; jj < 8; jj++)
                VsT[(q * 8 + jj) * 64 + row] = tb[jj];
        }
        if (kt + 1 < cS / 64) {
            const int row = tid >> 1, q = tid & 1;
            __builtin_prefetch(Kp + (size_t)((kt + 1) * 64 + row) * cDK + q * 32, 0, 1);
            __builtin_prefetch(Vp + (size_t)((kt + 1) * 64 + row) * cDK + q * 32, 0, 1);
        }
        async_wait();
        __syncthreads();

        // scores: 16q x 64k per wave = 4 C-frags, 2 WMMAs each (dk = 2x32)
        v8f sc[4];
        #pragma unroll
        for (int nt = 0; nt < 4; nt++) {
            v8f c = z;
            #pragma unroll
            for (int j = 0; j < 2; j++) {
                const v16bf bk = load_bfrag(Ks + (nt * 16 + lo) * 64 + j * 32, half);
                c = __builtin_amdgcn_wmma_f32_16x16x32_bf16(
                    false, aq[j], false, bk, (short)0, c, false, false);
            }
            sc[nt] = c;
        }

        // scale 1/sqrt(64) then mask
        #pragma unroll
        for (int nt = 0; nt < 4; nt++)
            #pragma unroll
            for (int r = 0; r < 8; r++) {
                const int qa = qw + r + half * 8;
                const int ka = kt * 64 + nt * 16 + lo;
                float v = sc[nt][r] * 0.125f;
                if (mp[(size_t)qa * cS + ka] == 0) v = -1e9f;
                sc[nt][r] = v;
            }

        // online softmax: row max over 64 keys (4 frags + 16-lane group xor)
        float alpha[8];
        #pragma unroll
        for (int r = 0; r < 8; r++) {
            float tmx = fmaxf(fmaxf(sc[0][r], sc[1][r]),
                              fmaxf(sc[2][r], sc[3][r]));
            #pragma unroll
            for (int off = 8; off >= 1; off >>= 1)
                tmx = fmaxf(tmx, __shfl_xor(tmx, off, 32));
            const float nm = fmaxf(mrow[r], tmx);
            alpha[r] = __expf(mrow[r] - nm);
            mrow[r] = nm;
        }

        // P = exp(s - m): write bf16 to per-wave LDS pad for layout change
        float ts[8];
        #pragma unroll
        for (int r = 0; r < 8; r++) ts[r] = 0.f;
        #pragma unroll
        for (int nt = 0; nt < 4; nt++)
            #pragma unroll
            for (int r = 0; r < 8; r++) {
                const float p = __expf(sc[nt][r] - mrow[r]);
                ts[r] += p;
                Ps[wid][(r + half * 8) * 64 + nt * 16 + lo] = (__bf16)p;
            }
        #pragma unroll
        for (int r = 0; r < 8; r++) {
            float t = ts[r];
            #pragma unroll
            for (int off = 8; off >= 1; off >>= 1)
                t += __shfl_xor(t, off, 32);
            lrow[r] = lrow[r] * alpha[r] + t;
        }
        // rescale running output
        #pragma unroll
        for (int j = 0; j < 4; j++)
            #pragma unroll
            for (int r = 0; r < 8; r++)
                o[j][r] *= alpha[r];
        __syncthreads();

        // P as A-frags (16q x 64keys, contract over keys in 2 chunks of 32)
        v16bf pa[2];
        #pragma unroll
        for (int j = 0; j < 2; j++) {
            const __bf16* pp = Ps[wid] + lo * 64 + j * 32 + half * 8;
            ((uint4*)&pa[j])[0] = *(const uint4*)(pp);
            ((uint4*)&pa[j])[1] = *(const uint4*)(pp + 16);
        }
        // O += P * V : 4 dk tiles x 2 key chunks (VsT is key-contig)
        #pragma unroll
        for (int dt = 0; dt < 4; dt++) {
            #pragma unroll
            for (int j = 0; j < 2; j++) {
                const v16bf vb = load_bfrag(VsT + (dt * 16 + lo) * 64 + j * 32, half);
                o[dt] = __builtin_amdgcn_wmma_f32_16x16x32_bf16(
                    false, pa[j], false, vb, (short)0, o[dt], false, false);
            }
        }
        __syncthreads();
    }

    // normalize and scatter into X[B,S,D] (col = h*64 + dk)
    #pragma unroll
    for (int dt = 0; dt < 4; dt++)
        #pragma unroll
        for (int r = 0; r < 8; r++) {
            const int qa = qw + r + half * 8;
            const int col = h * cDK + dt * 16 + lo;
            const float v = o[dt][r] / lrow[r];
            X[((size_t)(b * cS + qa)) * cD + col] = (__bf16)v;
        }
}

// ---------------------------------------------------------------------------
extern "C" void kernel_launch(void* const* d_in, const int* in_sizes, int n_in,
                              void* d_out, int out_size, void* d_ws, size_t ws_size,
                              hipStream_t stream)
{
    const float* k  = (const float*)d_in[0];
    const float* q  = (const float*)d_in[1];
    const float* v  = (const float*)d_in[2];
    const int* mask = (const int*)d_in[3];
    const float* Wq = (const float*)d_in[4];
    const float* bq = (const float*)d_in[5];
    const float* Wk = (const float*)d_in[6];
    const float* bk = (const float*)d_in[7];
    const float* Wv = (const float*)d_in[8];
    const float* bv = (const float*)d_in[9];
    const float* Wo = (const float*)d_in[10];
    const float* bo = (const float*)d_in[11];
    const float* a2 = (const float*)d_in[12];
    const float* b2 = (const float*)d_in[13];
    float* out = (float*)d_out;

    char* ws = (char*)d_ws;
    const size_t SZ_BF = (size_t)cM * cD * sizeof(__bf16);  // 8 MiB
    const size_t SZ_F  = (size_t)cM * cD * sizeof(float);   // 16 MiB
    const size_t SZ_W  = (size_t)cD * cD * sizeof(__bf16);  // 2 MiB

    size_t off = 0;
    __bf16* kn  = (__bf16*)(ws + off); off += SZ_BF;
    __bf16* qn  = (__bf16*)(ws + off); off += SZ_BF;
    __bf16* vn  = (__bf16*)(ws + off); off += SZ_BF;
    float*  vnf = (float*) (ws + off); off += SZ_F;
    __bf16* Wqb = (__bf16*)(ws + off); off += SZ_W;
    __bf16* Wkb = (__bf16*)(ws + off); off += SZ_W;
    __bf16* Wvb = (__bf16*)(ws + off); off += SZ_W;
    __bf16* Wob = (__bf16*)(ws + off); off += SZ_W;
    __bf16* Qb  = (__bf16*)(ws + off); off += SZ_BF;
    __bf16* Kb  = (__bf16*)(ws + off); off += SZ_BF;
    __bf16* Vb  = (__bf16*)(ws + off); off += SZ_BF;
    __bf16* Xb  = (__bf16*)(ws + off); off += SZ_BF;

    // 1) pre-norm all three streams (bf16), fp32 copy of vn for residual
    ln_kernel<<<dim3(cM), 256, 0, stream>>>(k, a2, b2, kn, nullptr);
    ln_kernel<<<dim3(cM), 256, 0, stream>>>(q, a2, b2, qn, nullptr);
    ln_kernel<<<dim3(cM), 256, 0, stream>>>(v, a2, b2, vn, vnf);

    // 2) weights -> bf16
    const int wblocks = (cD * cD) / (256 * 4);
    cvt_bf16<<<dim3(wblocks), 256, 0, stream>>>(Wq, Wqb);
    cvt_bf16<<<dim3(wblocks), 256, 0, stream>>>(Wk, Wkb);
    cvt_bf16<<<dim3(wblocks), 256, 0, stream>>>(Wv, Wvb);
    cvt_bf16<<<dim3(wblocks), 256, 0, stream>>>(Wo, Wob);

    // 3) projections (reference quirk: kn->query, qn->key, vn->value)
    dim3 ggrid(cM / 128, cD / 64);
    gemm16_qkv<<<ggrid, 256, 0, stream>>>(kn, Wqb, bq, Qb);
    gemm16_qkv<<<ggrid, 256, 0, stream>>>(qn, Wkb, bk, Kb);
    gemm16_qkv<<<ggrid, 256, 0, stream>>>(vn, Wvb, bv, Vb);

    // 4) fused masked attention (flash-style, online softmax)
    attn_fused<<<dim3(cS / 64, cH, cB), 128, 0, stream>>>(Qb, Kb, Vb, mask, Xb);

    // 5) output projection + bias + residual (normalized v), fp32 out
    gemm16_out<<<ggrid, 256, 0, stream>>>(Xb, Wob, bo, vnf, out);
}
